// AttentionTCN_GNN_15341623181912
// MI455X (gfx1250) — matrix operand order repaired and checked by reading, not compiled, not run on hardware
//
#include <hip/hip_runtime.h>
#include <math.h>

// ---------------- problem dims ----------------
#define Bz     8
#define Tt     64
#define Nn     200
#define FIN    16
#define Hh     64
#define NHEADS 2
#define Dd     32          // Hh / NHEADS
#define Ll     3
#define Kk     3
#define Ee     800
#define EMBe   10
#define ETOT   1000        // Ee + Nn (self loops)
#define BT     512         // Bz*Tt
#define BN     1600        // Bz*Nn
#define ROWS   102400      // BN*Tt == BT*Nn
#define ACTSZ  6553600     // ROWS*Hh

// padded adjacency: 200x200 stored in 208x224 (K multiple of 32, rows to tile edge)
#define ADJ_LDA 224
#define ADJ_RP  208

typedef __attribute__((ext_vector_type(16))) _Float16 v16h;
typedef __attribute__((ext_vector_type(8)))  float    v8f;

#define GF_ACC  1
#define GF_RELU 2

// ------------------------------------------------------------------
// WMMA GEMM, one wave computes a 16x64 strip of C = A(MxK)@B(KxN).
// fp32 in memory, f16 operands in-register, fp32 accumulate.
// BRANCHLESS inner loop: caller guarantees
//   - K % 32 == 0
//   - A has ceil(M/16)*16 rows of lda floats allocated (pad rows readable)
//   - B has K rows of ldb floats readable (overreads must be harmless:
//     either real zeros on the A side, or in-workspace garbage * 0)
//   - column strip tn..tn+63 valid in B/C (all our GEMMs have N==64)
// flags: GF_ACC -> C += result ; GF_RELU -> relu epilogue. bias may be null.
// Batched over blockIdx.z with element strides sA/sB/sC.
// ------------------------------------------------------------------
__global__ __launch_bounds__(32) void gemm16x64(
    const float* __restrict__ A, const float* __restrict__ B,
    const float* __restrict__ bias, float* __restrict__ C,
    int M, int K, int flags, int lda, int ldb, int ldc,
    long long sA, long long sB, long long sC)
{
  const int lane = threadIdx.x & 31;
  const int half = lane >> 4;          // which 16-lane half
  const int idx  = lane & 15;
  const int tm = blockIdx.y << 4;
  const int tn = blockIdx.x << 6;      // 64-wide output strip
  const float* Ab = A + (long long)blockIdx.z * sA;
  const float* Bb = B + (long long)blockIdx.z * sB;
  float*       Cb = C + (long long)blockIdx.z * sC;

  const float* Arow = Ab + (size_t)(tm + idx) * lda;

  v8f acc0 = {}, acc1 = {}, acc2 = {}, acc3 = {};
  for (int kk = 0; kk < K; kk += 32) {
    // A 16x32 f16 layout: lane(half,m): elems 0..15 -> K = kk+half*16+e
    v16h va;
#pragma unroll
    for (int e = 0; e < 8; ++e) {
      va[e]     = (_Float16)Arow[kk + half * 16 + e];
      va[e + 8] = (_Float16)Arow[kk + half * 16 + 8 + e];
    }
    // B 32x16 f16 layout: lanes 0-15 hold K = kk+0..15, lanes 16-31 K = kk+16..31
    const float* Bk = Bb + (size_t)(kk + half * 16) * ldb + tn + idx;
    v16h vb0, vb1, vb2, vb3;
#pragma unroll
    for (int e = 0; e < 16; ++e) {
      const float* Br = Bk + (size_t)e * ldb;
      vb0[e] = (_Float16)Br[0];
      vb1[e] = (_Float16)Br[16];
      vb2[e] = (_Float16)Br[32];
      vb3[e] = (_Float16)Br[48];
    }
    if (kk + 32 < K) __builtin_prefetch(Arow + kk + 32, 0, 1);
    acc0 = __builtin_amdgcn_wmma_f32_16x16x32_f16(false, va, false, vb0,
                                                  (short)0, acc0, false, false);
    acc1 = __builtin_amdgcn_wmma_f32_16x16x32_f16(false, va, false, vb1,
                                                  (short)0, acc1, false, false);
    acc2 = __builtin_amdgcn_wmma_f32_16x16x32_f16(false, va, false, vb2,
                                                  (short)0, acc2, false, false);
    acc3 = __builtin_amdgcn_wmma_f32_16x16x32_f16(false, va, false, vb3,
                                                  (short)0, acc3, false, false);
  }

  // epilogue (only place with bounds checks; all WMMAs already executed)
  const float bv0 = bias ? bias[tn + idx]      : 0.0f;
  const float bv1 = bias ? bias[tn + 16 + idx] : 0.0f;
  const float bv2 = bias ? bias[tn + 32 + idx] : 0.0f;
  const float bv3 = bias ? bias[tn + 48 + idx] : 0.0f;
#pragma unroll
  for (int r = 0; r < 8; ++r) {
    int rr = tm + half * 8 + r;        // C layout: VGPR r -> M = 8*half + r
    if (rr < M) {
      size_t o = (size_t)rr * ldc + tn + idx;
      float v0 = acc0[r] + bv0;
      float v1 = acc1[r] + bv1;
      float v2 = acc2[r] + bv2;
      float v3 = acc3[r] + bv3;
      if (flags & GF_ACC) {
        v0 += Cb[o]; v1 += Cb[o + 16]; v2 += Cb[o + 32]; v3 += Cb[o + 48];
      }
      if (flags & GF_RELU) {
        v0 = v0 > 0.0f ? v0 : 0.0f;
        v1 = v1 > 0.0f ? v1 : 0.0f;
        v2 = v2 > 0.0f ? v2 : 0.0f;
        v3 = v3 > 0.0f ? v3 : 0.0f;
      }
      Cb[o] = v0; Cb[o + 16] = v1; Cb[o + 32] = v2; Cb[o + 48] = v3;
    }
  }
}

// ---------------- utility / elementwise kernels ----------------
__global__ void k_zero(float* __restrict__ p, long long n) {
  long long i = (long long)blockIdx.x * blockDim.x + threadIdx.x;
  if (i < n) p[i] = 0.0f;
}

// pack x[(b t n),16] into zero-padded [(b t n),32]
__global__ void k_padx(const float* __restrict__ x, float* __restrict__ xp) {
  size_t s = (size_t)blockIdx.x * blockDim.x + threadIdx.x;
  if (s >= (size_t)ROWS * 32) return;
  int c = (int)(s % 32);
  size_t r = s / 32;
  xp[s] = (c < FIN) ? x[r * FIN + c] : 0.0f;
}

// pad enc_w [16,64] -> [32,64] (zeros below)
__global__ void k_prepenc(const float* __restrict__ w, float* __restrict__ wp) {
  int tid = blockIdx.x * blockDim.x + threadIdx.x;
  if (tid >= 32 * Hh) return;
  int k = tid / Hh;
  wp[tid] = (k < FIN) ? w[tid] : 0.0f;
}

__global__ void k_srcdst(const int* __restrict__ ei, int* __restrict__ src,
                         int* __restrict__ dst) {
  int e = blockIdx.x * blockDim.x + threadIdx.x;
  if (e < ETOT) {
    if (e < Ee) { src[e] = ei[e]; dst[e] = ei[Ee + e]; }
    else        { src[e] = e - Ee; dst[e] = e - Ee; }   // self loops
  }
}

// single-thread counting sort -> CSR grouped by dst (deterministic)
__global__ void k_csr(const int* __restrict__ dst, int* __restrict__ off,
                      int* __restrict__ csr) {
  if (blockIdx.x == 0 && threadIdx.x == 0) {
    int cnt[Nn];
    for (int i = 0; i < Nn; ++i) cnt[i] = 0;
    for (int e = 0; e < ETOT; ++e) cnt[dst[e]]++;
    off[0] = 0;
    for (int i = 0; i < Nn; ++i) off[i + 1] = off[i] + cnt[i];
    for (int i = 0; i < Nn; ++i) cnt[i] = off[i];
    for (int e = 0; e < ETOT; ++e) csr[cnt[dst[e]]++] = e;
  }
}

// adj[n,m] = softmax_m( relu( adj_src[n] . adj_tgt[m] ) ), written with lda=ADJ_LDA
__global__ __launch_bounds__(256) void k_adj(const float* __restrict__ as,
                                             const float* __restrict__ at,
                                             float* __restrict__ adj) {
  __shared__ float red[256];
  int n = blockIdx.x, tid = threadIdx.x;
  float logit = 0.0f;
  if (tid < Nn) {
    float acc = 0.0f;
    for (int e = 0; e < EMBe; ++e) acc += as[n * EMBe + e] * at[tid * EMBe + e];
    logit = acc > 0.0f ? acc : 0.0f;
  }
  red[tid] = (tid < Nn) ? logit : -1e30f;
  __syncthreads();
  for (int st = 128; st > 0; st >>= 1) {
    if (tid < st) red[tid] = fmaxf(red[tid], red[tid + st]);
    __syncthreads();
  }
  float mx = red[0];
  __syncthreads();
  float ex = (tid < Nn) ? expf(logit - mx) : 0.0f;
  red[tid] = ex;
  __syncthreads();
  for (int st = 128; st > 0; st >>= 1) {
    if (tid < st) red[tid] += red[tid + st];
    __syncthreads();
  }
  if (tid < Nn) adj[n * ADJ_LDA + tid] = ex / red[0];
}

// weight transforms: wct[l][(k*H+i)][o] = tcn_w[l][o][i][k]
//                    swt[l][i][o]       = skip_w[l][o][i]
__global__ void k_prepw(const float* __restrict__ tw, const float* __restrict__ sw,
                        float* __restrict__ wct, float* __restrict__ swt) {
  int tid = blockIdx.x * blockDim.x + threadIdx.x;
  const int NW = Ll * Kk * Hh * Hh;
  if (tid < NW) {
    int l  = tid / (Kk * Hh * Hh);
    int r  = tid % (Kk * Hh * Hh);
    int k  = r / (Hh * Hh);
    int i  = (r % (Hh * Hh)) / Hh;
    int oo = r % Hh;
    wct[tid] = tw[(((size_t)l * Hh + oo) * Hh + i) * Kk + k];
  }
  const int NS = Ll * Hh * Hh;
  if (tid < NS) {
    int l  = tid / (Hh * Hh);
    int i  = (tid % (Hh * Hh)) / Hh;
    int oo = tid % Hh;
    swt[tid] = sw[((size_t)l * Hh + oo) * Hh + i];
  }
}

// encoder epilogue: (b t n f) -> canonical (b n t f), + station_emb[n]
__global__ void k_enc_post(const float* __restrict__ enc, const float* __restrict__ emb,
                           float* __restrict__ act) {
  size_t s = (size_t)blockIdx.x * blockDim.x + threadIdx.x;
  if (s >= (size_t)ACTSZ) return;
  int f = (int)(s % Hh);
  int n = (int)((s / Hh) % Nn);
  int t = (int)((s / ((size_t)Hh * Nn)) % Tt);
  int b = (int)(s / ((size_t)Hh * Nn * Tt));
  act[(((size_t)(b * Nn + n)) * Tt + t) * Hh + f] = enc[s] + emb[n * Hh + f];
}

// canonical (b n t f) -> bt layout (b t n f)
__global__ void k_c2bt(const float* __restrict__ src, float* __restrict__ dst) {
  size_t s = (size_t)blockIdx.x * blockDim.x + threadIdx.x;
  if (s >= (size_t)ACTSZ) return;
  int f = (int)(s % Hh);
  int n = (int)((s / Hh) % Nn);
  int t = (int)((s / ((size_t)Hh * Nn)) % Tt);
  int b = (int)(s / ((size_t)Hh * Nn * Tt));
  dst[s] = src[(((size_t)(b * Nn + n)) * Tt + t) * Hh + f];
}

// bt layout (b t n f) -> canonical (b n t f)
__global__ void k_bt2c(const float* __restrict__ src, float* __restrict__ dst) {
  size_t s = (size_t)blockIdx.x * blockDim.x + threadIdx.x;
  if (s >= (size_t)ACTSZ) return;
  int f = (int)(s % Hh);
  int t = (int)((s / Hh) % Tt);
  int n = (int)((s / ((size_t)Hh * Tt)) % Nn);
  int b = (int)(s / ((size_t)Hh * Tt * Nn));
  dst[s] = src[(((size_t)(b * Tt + t)) * Nn + n) * Hh + f];
}

// im2col for dilated SAME conv, column = k*H + i
__global__ void k_im2col(const float* __restrict__ act, float* __restrict__ im, int dil) {
  size_t s = (size_t)blockIdx.x * blockDim.x + threadIdx.x;
  if (s >= (size_t)ROWS * (Kk * Hh)) return;
  int    c  = (int)(s % (Kk * Hh));
  size_t r  = s / (Kk * Hh);
  int    t  = (int)(r % Tt);
  size_t bn = r / Tt;
  int k = c / Hh, i = c % Hh;
  int ts = t + (k - 1) * dil;
  im[s] = (ts >= 0 && ts < Tt) ? act[(bn * Tt + ts) * Hh + i] : 0.0f;
}

// GATv2 edge logits: leaky_relu(xl[src]+xr[dst], 0.2) . a[h]
__global__ void k_logits(const float* __restrict__ xl, const float* __restrict__ xr,
                         const float* __restrict__ av, const int* __restrict__ src,
                         const int* __restrict__ dst, float* __restrict__ lg) {
  size_t tid = (size_t)blockIdx.x * blockDim.x + threadIdx.x;
  if (tid >= (size_t)ETOT * BT * NHEADS) return;
  int h  = (int)(tid % NHEADS);
  int bt = (int)((tid / NHEADS) % BT);
  int e  = (int)(tid / ((size_t)NHEADS * BT));
  int b = bt / Tt, t = bt % Tt;
  const float* a_ = av + h * Dd;
  const float* pl = xl + (((size_t)(b * Nn + src[e])) * Tt + t) * Hh + h * Dd;
  const float* pr = xr + (((size_t)(b * Nn + dst[e])) * Tt + t) * Hh + h * Dd;
  float acc = 0.0f;
  for (int d2 = 0; d2 < Dd; ++d2) {
    float v = pl[d2] + pr[d2];
    v = v > 0.0f ? v : 0.2f * v;
    acc += v * a_[d2];
  }
  lg[tid] = acc;                       // layout (e*BT+bt)*NHEADS + h
}

// per (dst-node, bt, head): segment max + sum(exp)
__global__ void k_segsoft(const float* __restrict__ lg, const int* __restrict__ off,
                          const int* __restrict__ csr, float* __restrict__ mb,
                          float* __restrict__ db) {
  size_t tid = (size_t)blockIdx.x * blockDim.x + threadIdx.x;
  if (tid >= (size_t)Nn * BT * NHEADS) return;
  int h  = (int)(tid % NHEADS);
  int bt = (int)((tid / NHEADS) % BT);
  int n  = (int)(tid / ((size_t)NHEADS * BT));
  int s0 = off[n], s1 = off[n + 1];
  float m = -1e30f;
  for (int j = s0; j < s1; ++j)
    m = fmaxf(m, lg[((size_t)csr[j] * BT + bt) * NHEADS + h]);
  float dsum = 0.0f;
  for (int j = s0; j < s1; ++j)
    dsum += expf(lg[((size_t)csr[j] * BT + bt) * NHEADS + h] - m);
  mb[tid] = m;
  db[tid] = dsum;
}

// aggregate alpha-weighted xl[src] into gat[(bt),n,f] (+bias)
__global__ void k_gatmsg(const float* __restrict__ xl, const float* __restrict__ lg,
                         const float* __restrict__ mb, const float* __restrict__ db,
                         const int* __restrict__ src, const int* __restrict__ off,
                         const int* __restrict__ csr, const float* __restrict__ bias,
                         float* __restrict__ gat) {
  size_t tid = (size_t)blockIdx.x * blockDim.x + threadIdx.x;
  if (tid >= (size_t)ACTSZ) return;
  int f  = (int)(tid % Hh);
  int n  = (int)((tid / Hh) % Nn);
  int bt = (int)(tid / ((size_t)Hh * Nn));
  int h = f / Dd;
  int b = bt / Tt, t = bt % Tt;
  size_t mi = ((size_t)n * BT + bt) * NHEADS + h;
  float m  = mb[mi];
  float dn = db[mi] + 1e-16f;
  float acc = 0.0f;
  int s1 = off[n + 1];
  for (int j = off[n]; j < s1; ++j) {
    int e = csr[j];
    float w = expf(lg[((size_t)e * BT + bt) * NHEADS + h] - m) / dn;
    acc += xl[(((size_t)(b * Nn + src[e])) * Tt + t) * Hh + f] * w;
  }
  gat[tid] = acc + bias[f];
}

// batchnorm stats: one block per channel, deterministic tree reduce
__global__ __launch_bounds__(256) void k_bnstat(const float* __restrict__ act,
                                                float* __restrict__ stats) {
  __shared__ float s1[256], s2[256];
  int c = blockIdx.x, tid = threadIdx.x;
  float a = 0.0f, b = 0.0f;
  for (int r = tid; r < ROWS; r += 256) {
    float v = act[(size_t)r * Hh + c];
    a += v; b += v * v;
  }
  s1[tid] = a; s2[tid] = b;
  __syncthreads();
  for (int st = 128; st > 0; st >>= 1) {
    if (tid < st) { s1[tid] += s1[tid + st]; s2[tid] += s2[tid + st]; }
    __syncthreads();
  }
  if (tid == 0) {
    float mu = s1[0] / (float)ROWS;
    stats[c]      = mu;
    stats[Hh + c] = s2[0] / (float)ROWS - mu * mu;
  }
}

__global__ void k_bnapply(float* __restrict__ act, const float* __restrict__ stats,
                          const float* __restrict__ g, const float* __restrict__ bb) {
  size_t s = (size_t)blockIdx.x * blockDim.x + threadIdx.x;
  if (s >= (size_t)ACTSZ) return;
  int c = (int)(s % Hh);
  float mu = stats[c], var = stats[Hh + c];
  act[s] = (act[s] - mu) * rsqrtf(var + 1e-5f) * g[c] + bb[c];
}

// head: res = skip + act (canonical); y0 = res.ow0, y1 = softplus(res.ow1)
__global__ void k_out(const float* __restrict__ skip, const float* __restrict__ act,
                      const float* __restrict__ ow, const float* __restrict__ ob,
                      float* __restrict__ out) {
  int s = blockIdx.x * blockDim.x + threadIdx.x;
  if (s >= ROWS) return;
  int n = s % Nn;
  int t = (s / Nn) % Tt;
  int b = s / (Nn * Tt);
  size_t row = ((size_t)(b * Nn + n)) * Tt + t;
  float a0 = 0.0f, a1 = 0.0f;
  for (int f = 0; f < Hh; ++f) {
    float r = skip[row * Hh + f] + act[row * Hh + f];
    a0 += r * ow[f * 2 + 0];
    a1 += r * ow[f * 2 + 1];
  }
  a0 += ob[0]; a1 += ob[1];
  float sp = (a1 > 20.0f) ? a1 : log1pf(expf(a1));
  out[(size_t)s * 2 + 0] = a0;
  out[(size_t)s * 2 + 1] = sp;
}

// ------------------------------------------------------------------
extern "C" void kernel_launch(void* const* d_in, const int* in_sizes, int n_in,
                              void* d_out, int out_size, void* d_ws, size_t ws_size,
                              hipStream_t stream) {
  const float* x       = (const float*)d_in[0];
  const int*   eidx    = (const int*)  d_in[1];
  const float* enc_w   = (const float*)d_in[2];
  const float* enc_b   = (const float*)d_in[3];
  const float* st_emb  = (const float*)d_in[4];
  const float* adj_src = (const float*)d_in[5];
  const float* adj_tgt = (const float*)d_in[6];
  const float* tcn_w   = (const float*)d_in[7];
  const float* tcn_b   = (const float*)d_in[8];
  const float* skip_w  = (const float*)d_in[9];
  const float* skip_b  = (const float*)d_in[10];
  const float* gat_wl  = (const float*)d_in[11];
  const float* gat_wr  = (const float*)d_in[12];
  const float* gat_a   = (const float*)d_in[13];
  const float* gat_b   = (const float*)d_in[14];
  const float* dense_w = (const float*)d_in[15];
  const float* dense_b = (const float*)d_in[16];
  const float* fus_w   = (const float*)d_in[17];
  const float* fus_b   = (const float*)d_in[18];
  const float* bn_g    = (const float*)d_in[19];
  const float* bn_b    = (const float*)d_in[20];
  const float* out_w   = (const float*)d_in[21];
  const float* out_b   = (const float*)d_in[22];

  // -------- workspace layout (~300 MB of f32 + small int region) --------
  float* F = (float*)d_ws;
  size_t off = 0;
  auto alloc = [&](size_t n) { float* p = F + off; off += n; return p; };
  float* ACT  = alloc(ACTSZ);         // canonical (b n t f) activations
  float* HB   = alloc(ACTSZ);         // conv output (canonical)
  float* XL   = alloc(ACTSZ);         // GAT src projection (canonical)
  float* XR   = alloc(ACTSZ);         // GAT dst projection (canonical)
  float* SK   = alloc(ACTSZ);         // skip_sum (canonical)
  float* TBT  = alloc(ACTSZ);         // xs in (b t n f) layout
  float* FUS  = alloc(ACTSZ);         // fusion output (bt layout); also enc tmp
  float* R    = alloc(3 * (size_t)ACTSZ); // aliased: im2col | gat/den1/den2
  float* GAT  = R;
  float* DEN1 = R + ACTSZ;
  float* DEN2 = R + 2 * (size_t)ACTSZ;
  float* IM   = R;                    // im2col (ROWS*192 == 3*ACTSZ) — dead before GAT/DEN
  float* XPAD = alloc((size_t)ROWS * 32);      // x zero-padded to K=32
  float* LG   = alloc((size_t)ETOT * BT * NHEADS);
  float* MB   = alloc((size_t)Nn * BT * NHEADS);
  float* DB   = alloc((size_t)Nn * BT * NHEADS);
  float* ADJP = alloc((size_t)ADJ_RP * ADJ_LDA); // padded adjacency, zero-filled
  float* WCT  = alloc((size_t)Ll * Kk * Hh * Hh);
  float* SWT  = alloc((size_t)Ll * Hh * Hh);
  float* WENC = alloc(32 * Hh);                // enc_w padded to K=32
  float* ST   = alloc(2 * Hh);
  int* I   = (int*)(F + off);
  int* SRC = I;
  int* DST = I + ETOT;
  int* OFF = I + 2 * ETOT;
  int* CSR = OFF + Nn + 8;

  dim3 b256(256);
  auto nb = [](size_t n) { return dim3((unsigned)((n + 255) / 256)); };
  // N is always 64 -> one 16x64 strip per M-tile
  auto gemm = [&](const float* A, const float* B_, const float* bias, float* C,
                  int M, int K, int flags, int lda, int ldb, int ldc, int batch,
                  long long sA, long long sB, long long sC) {
    dim3 g(1, (M + 15) / 16, batch);
    gemm16x64<<<g, dim3(32), 0, stream>>>(A, B_, bias, C, M, K, flags,
                                          lda, ldb, ldc, sA, sB, sC);
  };

  // -------- setup --------
  k_zero   <<<nb(ACTSZ), b256, 0, stream>>>(SK, (long long)ACTSZ);
  k_zero   <<<nb(ADJ_RP * ADJ_LDA), b256, 0, stream>>>(ADJP, ADJ_RP * ADJ_LDA);
  k_srcdst <<<nb(ETOT),  b256, 0, stream>>>(eidx, SRC, DST);
  k_csr    <<<1, 1, 0, stream>>>(DST, OFF, CSR);
  k_adj    <<<Nn, 256, 0, stream>>>(adj_src, adj_tgt, ADJP);
  k_prepw  <<<nb(Ll * Kk * Hh * Hh), b256, 0, stream>>>(tcn_w, skip_w, WCT, SWT);
  k_prepenc<<<nb(32 * Hh), b256, 0, stream>>>(enc_w, WENC);
  k_padx   <<<nb((size_t)ROWS * 32), b256, 0, stream>>>(x, XPAD);

  // -------- encoder: xpad[(b t n),32] @ wenc[32,64] + enc_b --------
  gemm(XPAD, WENC, enc_b, FUS, ROWS, 32, 0, 32, Hh, Hh, 1, 0, 0, 0);
  k_enc_post<<<nb(ACTSZ), b256, 0, stream>>>(FUS, st_emb, ACT);

  // -------- layers --------
  for (int l = 0; l < Ll; ++l) {
    int dil = 1 << l;
    // TCN dilated conv as im2col + GEMM (+bias, relu)
    k_im2col<<<nb((size_t)ROWS * Kk * Hh), b256, 0, stream>>>(ACT, IM, dil);
    gemm(IM, WCT + (size_t)l * Kk * Hh * Hh, tcn_b + l * Hh, HB,
         ROWS, Kk * Hh, GF_RELU, Kk * Hh, Hh, Hh, 1, 0, 0, 0);
    // skip_sum += h @ skip_w^T + skip_b
    gemm(HB, SWT + (size_t)l * Hh * Hh, skip_b + l * Hh, SK,
         ROWS, Hh, GF_ACC, Hh, Hh, Hh, 1, 0, 0, 0);
    // GATv2 projections
    gemm(HB, gat_wl + (size_t)l * Hh * Hh, nullptr, XL,
         ROWS, Hh, 0, Hh, Hh, Hh, 1, 0, 0, 0);
    gemm(HB, gat_wr + (size_t)l * Hh * Hh, nullptr, XR,
         ROWS, Hh, 0, Hh, Hh, Hh, 1, 0, 0, 0);
    // edge path
    k_logits <<<nb((size_t)ETOT * BT * NHEADS), b256, 0, stream>>>(
        XL, XR, gat_a + l * NHEADS * Dd, SRC, DST, LG);
    k_segsoft<<<nb((size_t)Nn * BT * NHEADS), b256, 0, stream>>>(LG, OFF, CSR, MB, DB);
    // xs in bt layout for dense path
    k_c2bt<<<nb(ACTSZ), b256, 0, stream>>>(HB, TBT);
    // GAT aggregation (overwrites im2col region — im2col is dead now)
    k_gatmsg<<<nb(ACTSZ), b256, 0, stream>>>(XL, LG, MB, DB, SRC, OFF, CSR,
                                             gat_b + l * Hh, GAT);
    // dense: adjP[200x224(z)] @ xs  batched over 512 (b,t) slices
    // (A pad columns are zero, so B overreads past row 200 contribute 0)
    gemm(ADJP, TBT, nullptr, DEN1, Nn, ADJ_LDA, 0, ADJ_LDA, Hh, Hh,
         BT, 0, (long long)Nn * Hh, (long long)Nn * Hh);
    gemm(DEN1, dense_w + (size_t)l * Hh * Hh, dense_b + l * Hh, DEN2,
         ROWS, Hh, 0, Hh, Hh, Hh, 1, 0, 0, 0);
    // fusion: concat(gat, dense) @ fus_w + fus_b, relu (two accumulating GEMMs)
    gemm(GAT,  fus_w + (size_t)l * 2 * Hh * Hh,           nullptr,        FUS,
         ROWS, Hh, 0, Hh, Hh, Hh, 1, 0, 0, 0);
    gemm(DEN2, fus_w + (size_t)l * 2 * Hh * Hh + Hh * Hh, fus_b + l * Hh, FUS,
         ROWS, Hh, GF_ACC | GF_RELU, Hh, Hh, Hh, 1, 0, 0, 0);
    // back to canonical + batchnorm (training-mode batch stats)
    k_bt2c   <<<nb(ACTSZ), b256, 0, stream>>>(FUS, ACT);
    k_bnstat <<<Hh, 256, 0, stream>>>(ACT, ST);
    k_bnapply<<<nb(ACTSZ), b256, 0, stream>>>(ACT, ST, bn_g + l * Hh, bn_b + l * Hh);
  }

  // -------- head --------
  k_out<<<nb(ROWS), b256, 0, stream>>>(SK, ACT, out_w, out_b, (float*)d_out);
}
